// Attention_5892695130739
// MI455X (gfx1250) — matrix operand order repaired and checked by reading, not compiled, not run on hardware
//
#include <hip/hip_runtime.h>
#include <hip/hip_bf16.h>

typedef __bf16 bf16_t;
typedef __attribute__((ext_vector_type(16))) __bf16 v16bf;
typedef __attribute__((ext_vector_type(8)))  __bf16 v8bf;
typedef __attribute__((ext_vector_type(8)))  float  v8f;

constexpr int BSZ = 2, SEQ = 2048, NH = 16, HD = 64;
constexpr int H = NH * HD;        // 1024
constexpr int M = BSZ * SEQ;      // 4096

// Padded LDS pitches (bank-conflict avoidance for 64x4B banks):
//   pitch 40 bf16 -> 20 banks/row: r*20 mod 64 distinct for r=0..15
//   pitch 72 bf16 -> 36 banks/row: r*36 mod 64 distinct for r=0..15
constexpr int PB = 40;   // B-tile / V-tile / pbuf pitch
constexpr int PK = 72;   // K-tile pitch

// ---------------------------------------------------------------------------
// Async global->LDS (CDNA5 GLOBAL_LOAD_ASYNC_TO_LDS_B128, ASYNCcnt) with
// sync fallback (regs + ds_store) that keeps the same pipelining shape.
// Builtin signature (from hipcc diagnostic): first param is
//   'int __attribute__((vector_size(16))) __device__(AS1) *'
// ---------------------------------------------------------------------------
#if defined(__has_builtin)
#if __has_builtin(__builtin_amdgcn_global_load_async_to_lds_b128) && \
    __has_builtin(__builtin_amdgcn_s_wait_asynccnt)
#define HAVE_ASYNC 1
#endif
#endif
#ifndef HAVE_ASYNC
#define HAVE_ASYNC 0
#endif

#if HAVE_ASYNC
typedef int v4i_vs __attribute__((vector_size(16)));
typedef __attribute__((address_space(1))) v4i_vs* as1_v4i;
typedef __attribute__((address_space(3))) v4i_vs* as3_v4i;
#endif

__device__ inline void async_cp16(const bf16_t* g, bf16_t* l) {
#if HAVE_ASYNC
    __builtin_amdgcn_global_load_async_to_lds_b128(
        (as1_v4i)(v4i_vs*)g, (as3_v4i)(v4i_vs*)l, 0, 0);
#else
    (void)g; (void)l;
#endif
}
__device__ inline void async_wait0() {
#if HAVE_ASYNC
    __builtin_amdgcn_s_wait_asynccnt(0);
#endif
}

// ---------------------------------------------------------------------------
// WMMA helpers (gfx1250 wave32 layouts)
// ---------------------------------------------------------------------------
__device__ inline v8f wmma_bf16(v16bf a, v16bf b, v8f c) {
    return __builtin_amdgcn_wmma_f32_16x16x32_bf16(
        false, a, false, b, (short)0, c, false, false);
}

__device__ inline v16bf concat8(v8bf lo, v8bf hi) {
    return __builtin_shufflevector(lo, hi, 0,1,2,3,4,5,6,7,8,9,10,11,12,13,14,15);
}

// A-fragment (16x32 bf16 MxK) from row-major memory. lane: m = lane&15.
// elems 0..15 -> K = k0+16*half+0..15
__device__ inline v16bf load_a_frag(const bf16_t* p, int ld, int m, int k0, int half) {
    const bf16_t* base = p + (size_t)m * ld + k0 + half * 16;
    return concat8(*(const v8bf*)base, *(const v8bf*)(base + 8));
}

// B-fragment (32x16 bf16 KxN): lane holds column n, 16 contiguous K values
// starting at k-offset 16*half. `p` points at element (k=16*half, n) already.
__device__ inline v16bf load_b16(const bf16_t* p) {
    return concat8(*(const v8bf*)p, *(const v8bf*)(p + 8));
}

__device__ inline float rmax16(float v) {
    v = fmaxf(v, __shfl_xor(v, 1, 16));
    v = fmaxf(v, __shfl_xor(v, 2, 16));
    v = fmaxf(v, __shfl_xor(v, 4, 16));
    v = fmaxf(v, __shfl_xor(v, 8, 16));
    return v;
}
__device__ inline float rsum16(float v) {
    v += __shfl_xor(v, 1, 16);
    v += __shfl_xor(v, 2, 16);
    v += __shfl_xor(v, 4, 16);
    v += __shfl_xor(v, 8, 16);
    return v;
}

// ---------------------------------------------------------------------------
// Conversion kernels
// ---------------------------------------------------------------------------
__global__ void cvt_bf16_kernel(const float* __restrict__ src, bf16_t* __restrict__ dst, int n) {
    int i = blockIdx.x * blockDim.x + threadIdx.x;
    int stride = gridDim.x * blockDim.x;
    for (; i < n; i += stride) dst[i] = (bf16_t)src[i];
}

__global__ void cvt_transpose_kernel(const float* __restrict__ W0, const float* __restrict__ W1,
                                     const float* __restrict__ W2, const float* __restrict__ W3,
                                     bf16_t* __restrict__ T0, bf16_t* __restrict__ T1,
                                     bf16_t* __restrict__ T2, bf16_t* __restrict__ T3) {
    int z = blockIdx.y;
    const float* W = (z == 0) ? W0 : (z == 1) ? W1 : (z == 2) ? W2 : W3;
    bf16_t* T = (z == 0) ? T0 : (z == 1) ? T1 : (z == 2) ? T2 : T3;
    int idx = blockIdx.x * blockDim.x + threadIdx.x;     // k*H + n, n fastest
    int k = idx >> 10, n = idx & (H - 1);
    T[(size_t)n * H + k] = (bf16_t)W[idx];
}

// ---------------------------------------------------------------------------
// QKV projection: block = 4 waves = 64x64 tile. B slab (64 n-rows x 32 k)
// double-buffered in LDS, shared by all waves; A register-direct + prefetch.
//   z=0 -> Q [B,h,S,d], z=1 -> K [B,h,S,d], z=2 -> Vt [B,h,d,S]
// ---------------------------------------------------------------------------
__global__ void qkv_gemm_kernel(const bf16_t* __restrict__ xb,
                                const bf16_t* __restrict__ Wqt, const bf16_t* __restrict__ Wkt,
                                const bf16_t* __restrict__ Wvt,
                                const float* __restrict__ bq, const float* __restrict__ bk,
                                const float* __restrict__ bv,
                                bf16_t* __restrict__ Q, bf16_t* __restrict__ Kh,
                                bf16_t* __restrict__ Vt) {
    __shared__ __align__(16) bf16_t ldsB[2][64][PB];

    int z = blockIdx.y;
    const bf16_t* Wt = (z == 0) ? Wqt : (z == 1) ? Wkt : Wvt;
    const float* bias = (z == 0) ? bq : (z == 1) ? bk : bv;

    int tid = threadIdx.x;
    int wave = tid >> 5, lane = tid & 31;
    int half = lane >> 4, lm = lane & 15;

    int mb = blockIdx.x >> 4;          // 64 m-blocks of 64 rows
    int nb = blockIdx.x & 15;          // 16 n-blocks of 64 cols
    int m0 = mb * 64 + wave * 16, n0 = nb * 64;

    // staging map: each thread stages 16 contiguous bf16 (two 16B chunks)
    int srow = tid >> 1;               // 0..63
    int skel = (tid & 1) * 16;         // 0 / 16

    v8bf st0, st1;
    // prologue: stage k-slab 0 into buffer 0
    {
        const bf16_t* src = Wt + (size_t)(n0 + srow) * H + skel;
        bf16_t* dst = &ldsB[0][srow][skel];
#if HAVE_ASYNC
        async_cp16(src, dst); async_cp16(src + 8, dst + 8);
        async_wait0();
#else
        st0 = *(const v8bf*)src; st1 = *(const v8bf*)(src + 8);
        *(v8bf*)dst = st0; *(v8bf*)(dst + 8) = st1;
#endif
    }
    __syncthreads();

    v16bf a_cur = load_a_frag(xb, H, m0 + lm, 0, half);
    v8f acc[4] = {};
    int cur = 0;
    for (int k0 = 0; k0 < H; k0 += 32) {
        int nxt = k0 + 32;
        bool havep = nxt < H;
        v16bf a_nxt;
        if (havep) {
            const bf16_t* src = Wt + (size_t)(n0 + srow) * H + nxt + skel;
            bf16_t* dst = &ldsB[cur ^ 1][srow][skel];
#if HAVE_ASYNC
            async_cp16(src, dst); async_cp16(src + 8, dst + 8);
#else
            st0 = *(const v8bf*)src; st1 = *(const v8bf*)(src + 8);
#endif
            a_nxt = load_a_frag(xb, H, m0 + lm, nxt, half);
        }
        v16bf bfrag[4];
#pragma unroll
        for (int t = 0; t < 4; ++t)
            bfrag[t] = load_b16(&ldsB[cur][t * 16 + lm][half * 16]);
#pragma unroll
        for (int t = 0; t < 4; ++t)
            acc[t] = wmma_bf16(a_cur, bfrag[t], acc[t]);
        if (havep) {
#if HAVE_ASYNC
            async_wait0();
#else
            bf16_t* dst = &ldsB[cur ^ 1][srow][skel];
            *(v8bf*)dst = st0; *(v8bf*)(dst + 8) = st1;
#endif
            __syncthreads();
            a_cur = a_nxt;
            cur ^= 1;
        }
    }

#pragma unroll
    for (int t = 0; t < 4; ++t) {
        int ncol = n0 + t * 16 + lm;
        float bn = bias[ncol];
        int h = ncol >> 6, d = ncol & 63;
#pragma unroll
        for (int j = 0; j < 8; ++j) {
            int mrow = m0 + j + 8 * half;
            int b = mrow >> 11, s = mrow & (SEQ - 1);
            float v = acc[t][j] + bn;
            if (z == 2)
                Vt[(((size_t)(b * NH + h) * HD + d) * SEQ) + s] = (bf16_t)v;
            else {
                bf16_t* o = (z == 0) ? Q : Kh;
                o[(((size_t)(b * NH + h) * SEQ + s) * HD) + d] = (bf16_t)v;
            }
        }
    }
}

// ---------------------------------------------------------------------------
// Flash attention: block = 4 waves = 64 q-rows of one (b,h). 32-key blocks.
// K (32x64) and Vt (64x32) slabs double-buffered in LDS, shared by 4 waves.
// ---------------------------------------------------------------------------
__global__ void attn_kernel(const bf16_t* __restrict__ Q, const bf16_t* __restrict__ K,
                            const bf16_t* __restrict__ Vt, bf16_t* __restrict__ ctxb) {
    __shared__ __align__(16) bf16_t ldsK[2][32][PK];
    __shared__ __align__(16) bf16_t ldsV[2][64][PB];
    __shared__ __align__(16) bf16_t pbuf[4][16][PB];

    int tid = threadIdx.x;
    int wave = tid >> 5, lane = tid & 31;
    int half = lane >> 4, lm = lane & 15;

    int bh = blockIdx.y;
    int b = bh >> 4, h = bh & 15;
    const bf16_t* Qp = Q + (size_t)bh * SEQ * HD;
    const bf16_t* Kp = K + (size_t)bh * SEQ * HD;
    const bf16_t* Vp = Vt + (size_t)bh * HD * SEQ;

    int q0 = blockIdx.x * 64 + wave * 16;

    // staging maps (each thread stages 16 bf16 for K and 16 bf16 for V)
    int krow = tid >> 2, kkel = (tid & 3) * 16;   // K: 32 rows x 64 elems
    int vrow = tid >> 1, vkel = (tid & 1) * 16;   // V: 64 rows x 32 elems

    v8bf sk0, sk1, sv0, sv1;
    // prologue: stage key-block 0 into buffer 0
    {
        const bf16_t* ks = Kp + (size_t)krow * HD + kkel;
        const bf16_t* vs = Vp + (size_t)vrow * SEQ + vkel;
        bf16_t* kd = &ldsK[0][krow][kkel];
        bf16_t* vd = &ldsV[0][vrow][vkel];
#if HAVE_ASYNC
        async_cp16(ks, kd); async_cp16(ks + 8, kd + 8);
        async_cp16(vs, vd); async_cp16(vs + 8, vd + 8);
        async_wait0();
#else
        sk0 = *(const v8bf*)ks; sk1 = *(const v8bf*)(ks + 8);
        sv0 = *(const v8bf*)vs; sv1 = *(const v8bf*)(vs + 8);
        *(v8bf*)kd = sk0; *(v8bf*)(kd + 8) = sk1;
        *(v8bf*)vd = sv0; *(v8bf*)(vd + 8) = sv1;
#endif
    }
    __syncthreads();

    v16bf qa0 = load_a_frag(Qp, HD, q0 + lm, 0, half);
    v16bf qa1 = load_a_frag(Qp, HD, q0 + lm, 32, half);

    v8f ctx[4] = {};
    float mst[8], lst[8];
#pragma unroll
    for (int j = 0; j < 8; ++j) { mst[j] = -__builtin_inff(); lst[j] = 0.f; }

    const float sc = 0.125f * 1.44269504088896340736f;  // 1/sqrt(64) * log2(e)

    int cur = 0;
    for (int kb = 0; kb < SEQ; kb += 32) {
        int kbn = kb + 32;
        bool havep = kbn < SEQ;
        if (havep) {
            const bf16_t* ks = Kp + (size_t)(kbn + krow) * HD + kkel;
            const bf16_t* vs = Vp + (size_t)vrow * SEQ + kbn + vkel;
            bf16_t* kd = &ldsK[cur ^ 1][krow][kkel];
            bf16_t* vd = &ldsV[cur ^ 1][vrow][vkel];
#if HAVE_ASYNC
            async_cp16(ks, kd); async_cp16(ks + 8, kd + 8);
            async_cp16(vs, vd); async_cp16(vs + 8, vd + 8);
#else
            sk0 = *(const v8bf*)ks; sk1 = *(const v8bf*)(ks + 8);
            sv0 = *(const v8bf*)vs; sv1 = *(const v8bf*)(vs + 8);
#endif
        }

        // --- scores: S0 (keys kb..kb+15), S1 (kb+16..kb+31) from ldsK[cur]
        v8f s0 = {}, s1 = {};
        {
            v16bf b00 = load_b16(&ldsK[cur][lm][16 * half]);
            v16bf b01 = load_b16(&ldsK[cur][lm][32 + 16 * half]);
            v16bf b10 = load_b16(&ldsK[cur][16 + lm][16 * half]);
            v16bf b11 = load_b16(&ldsK[cur][16 + lm][32 + 16 * half]);
            s0 = wmma_bf16(qa0, b00, s0);
            s0 = wmma_bf16(qa1, b01, s0);
            s1 = wmma_bf16(qa0, b10, s1);
            s1 = wmma_bf16(qa1, b11, s1);
        }

        // --- online softmax
        float p0[8], p1[8], alpha[8];
#pragma unroll
        for (int j = 0; j < 8; ++j) {
            float x0 = s0[j] * sc, x1 = s1[j] * sc;
            float mx = rmax16(fmaxf(x0, x1));
            float mnew = fmaxf(mst[j], mx);
            float a = exp2f(mst[j] - mnew);
            float e0 = exp2f(x0 - mnew);
            float e1 = exp2f(x1 - mnew);
            float rs = rsum16(e0 + e1);
            lst[j] = lst[j] * a + rs;
            mst[j] = mnew;
            alpha[j] = a;
            p0[j] = e0; p1[j] = e1;
        }
#pragma unroll
        for (int t = 0; t < 4; ++t)
#pragma unroll
            for (int j = 0; j < 8; ++j) ctx[t][j] *= alpha[j];

        // --- P: C-layout -> A-layout via per-wave LDS tile
#pragma unroll
        for (int j = 0; j < 8; ++j) {
            int m = j + 8 * half;
            pbuf[wave][m][lm]      = (bf16_t)p0[j];
            pbuf[wave][m][16 + lm] = (bf16_t)p1[j];
        }
        __syncthreads();
        v16bf pa;
        {
            const bf16_t* base = &pbuf[wave][lm][half * 16];
            pa = concat8(*(const v8bf*)base, *(const v8bf*)(base + 8));
        }

        // --- ctx += P @ V from ldsV[cur]
#pragma unroll
        for (int t = 0; t < 4; ++t) {
            v16bf vb = load_b16(&ldsV[cur][t * 16 + lm][16 * half]);
            ctx[t] = wmma_bf16(pa, vb, ctx[t]);
        }

        if (havep) {
#if HAVE_ASYNC
            async_wait0();
#else
            bf16_t* kd = &ldsK[cur ^ 1][krow][kkel];
            bf16_t* vd = &ldsV[cur ^ 1][vrow][vkel];
            *(v8bf*)kd = sk0; *(v8bf*)(kd + 8) = sk1;
            *(v8bf*)vd = sv0; *(v8bf*)(vd + 8) = sv1;
#endif
            __syncthreads();
            cur ^= 1;
        }
    }

    float inv[8];
#pragma unroll
    for (int j = 0; j < 8; ++j) inv[j] = 1.0f / lst[j];

#pragma unroll
    for (int t = 0; t < 4; ++t) {
        int ncol = h * HD + t * 16 + lm;
#pragma unroll
        for (int j = 0; j < 8; ++j) {
            int mrow = q0 + j + 8 * half;
            ctxb[((size_t)(b * SEQ + mrow) * H) + ncol] = (bf16_t)(ctx[t][j] * inv[j]);
        }
    }
}

// ---------------------------------------------------------------------------
// Output projection: out = ctxb @ Wo + bo (fp32), same LDS-staged structure.
// ---------------------------------------------------------------------------
__global__ void out_gemm_kernel(const bf16_t* __restrict__ ctxb, const bf16_t* __restrict__ Wot,
                                const float* __restrict__ bo, float* __restrict__ out) {
    __shared__ __align__(16) bf16_t ldsB[2][64][PB];

    int tid = threadIdx.x;
    int wave = tid >> 5, lane = tid & 31;
    int half = lane >> 4, lm = lane & 15;

    int mb = blockIdx.x >> 4, nb = blockIdx.x & 15;
    int m0 = mb * 64 + wave * 16, n0 = nb * 64;

    int srow = tid >> 1, skel = (tid & 1) * 16;

    v8bf st0, st1;
    {
        const bf16_t* src = Wot + (size_t)(n0 + srow) * H + skel;
        bf16_t* dst = &ldsB[0][srow][skel];
#if HAVE_ASYNC
        async_cp16(src, dst); async_cp16(src + 8, dst + 8);
        async_wait0();
#else
        st0 = *(const v8bf*)src; st1 = *(const v8bf*)(src + 8);
        *(v8bf*)dst = st0; *(v8bf*)(dst + 8) = st1;
#endif
    }
    __syncthreads();

    v16bf a_cur = load_a_frag(ctxb, H, m0 + lm, 0, half);
    v8f acc[4] = {};
    int cur = 0;
    for (int k0 = 0; k0 < H; k0 += 32) {
        int nxt = k0 + 32;
        bool havep = nxt < H;
        v16bf a_nxt;
        if (havep) {
            const bf16_t* src = Wot + (size_t)(n0 + srow) * H + nxt + skel;
            bf16_t* dst = &ldsB[cur ^ 1][srow][skel];
#if HAVE_ASYNC
            async_cp16(src, dst); async_cp16(src + 8, dst + 8);
#else
            st0 = *(const v8bf*)src; st1 = *(const v8bf*)(src + 8);
#endif
            a_nxt = load_a_frag(ctxb, H, m0 + lm, nxt, half);
        }
        v16bf bfrag[4];
#pragma unroll
        for (int t = 0; t < 4; ++t)
            bfrag[t] = load_b16(&ldsB[cur][t * 16 + lm][half * 16]);
#pragma unroll
        for (int t = 0; t < 4; ++t)
            acc[t] = wmma_bf16(a_cur, bfrag[t], acc[t]);
        if (havep) {
#if HAVE_ASYNC
            async_wait0();
#else
            bf16_t* dst = &ldsB[cur ^ 1][srow][skel];
            *(v8bf*)dst = st0; *(v8bf*)(dst + 8) = st1;
#endif
            __syncthreads();
            a_cur = a_nxt;
            cur ^= 1;
        }
    }

#pragma unroll
    for (int t = 0; t < 4; ++t) {
        int ncol = n0 + t * 16 + lm;
        float bn = bo[ncol];
#pragma unroll
        for (int j = 0; j < 8; ++j) {
            int mrow = m0 + j + 8 * half;
            out[(size_t)mrow * H + ncol] = acc[t][j] + bn;
        }
    }
}

// ---------------------------------------------------------------------------
extern "C" void kernel_launch(void* const* d_in, const int* in_sizes, int n_in,
                              void* d_out, int out_size, void* d_ws, size_t ws_size,
                              hipStream_t stream) {
    const float* x  = (const float*)d_in[0];
    const float* Wq = (const float*)d_in[1];
    const float* bq = (const float*)d_in[2];
    const float* Wk = (const float*)d_in[3];
    const float* bk = (const float*)d_in[4];
    const float* Wv = (const float*)d_in[5];
    const float* bv = (const float*)d_in[6];
    const float* Wo = (const float*)d_in[7];
    const float* bo = (const float*)d_in[8];
    float* out = (float*)d_out;

    char* ws = (char*)d_ws;
    const size_t MB = 1024ull * 1024ull;
    bf16_t* xb   = (bf16_t*)(ws + 0 * MB);    // [4096,1024]            8 MB
    bf16_t* Wqt  = (bf16_t*)(ws + 8 * MB);    // [1024,1024] transposed 2 MB
    bf16_t* Wkt  = (bf16_t*)(ws + 10 * MB);
    bf16_t* Wvt  = (bf16_t*)(ws + 12 * MB);
    bf16_t* Wot  = (bf16_t*)(ws + 14 * MB);
    bf16_t* Qb   = (bf16_t*)(ws + 16 * MB);   // [B,h,S,d]              8 MB
    bf16_t* Kb   = (bf16_t*)(ws + 24 * MB);   // [B,h,S,d]              8 MB
    bf16_t* Vtb  = (bf16_t*)(ws + 32 * MB);   // [B,h,d,S]              8 MB
    bf16_t* ctxb = (bf16_t*)(ws + 40 * MB);   // [B*S,H]                8 MB

    cvt_bf16_kernel<<<2048, 256, 0, stream>>>(x, xb, M * H);
    cvt_transpose_kernel<<<dim3(H * H / 256, 4), 256, 0, stream>>>(
        Wq, Wk, Wv, Wo, Wqt, Wkt, Wvt, Wot);
    qkv_gemm_kernel<<<dim3(1024, 3), 128, 0, stream>>>(
        xb, Wqt, Wkt, Wvt, bq, bk, bv, Qb, Kb, Vtb);
    attn_kernel<<<dim3(SEQ / 64, BSZ * NH), 128, 0, stream>>>(Qb, Kb, Vtb, ctxb);
    out_gemm_kernel<<<1024, 128, 0, stream>>>(ctxb, Wot, bo, out);
}
